// SigmaMoE_17205638988279
// MI455X (gfx1250) — matrix operand order repaired and checked by reading, not compiled, not run on hardware
//
#include <hip/hip_runtime.h>
#include <hip/hip_bf16.h>

// Problem constants (match reference)
#define BB 4
#define SEQ 4096
#define DD 1024
#define EE 16
#define HH 128
#define NTOK (BB * SEQ)

#define MTILE 32               // tokens per workgroup (2 x 16-row WMMA tiles)

typedef __attribute__((ext_vector_type(16))) __bf16 v16bf;
typedef __attribute__((ext_vector_type(8)))  float  v8f;

// LDS row strides (elements). Must be multiple of 8 (16B-aligned rows for
// b128 loads); +8 pad spreads banks for the row-strided fragment gathers.
#define XS_STRIDE 1032   // 32 rows x 1024 cols (+8 pad)
#define SC_STRIDE 136    // 32 rows x 128 cols  (+8 pad)

// ---------------------------------------------------------------------------
// Prep: convert fp32 weights to bf16 layouts that make WMMA B-fragments
// contiguous per lane:
//   selW [E][D]        (row e = expert_sel[e][:])           -> router B
//   keysT[E][H][D]     (= keys[e][d][h] transposed)         -> phase-A B
//   valsT[E][D][H]     (= values[e][h][n] transposed)       -> phase-B B
// ---------------------------------------------------------------------------
__global__ void prep_kernel(const float* __restrict__ wsel,
                            const float* __restrict__ keys,
                            const float* __restrict__ vals,
                            __bf16* __restrict__ selW,
                            __bf16* __restrict__ keysT,
                            __bf16* __restrict__ valsT) {
  long tid = (long)blockIdx.x * blockDim.x + threadIdx.x;
  long stride = (long)gridDim.x * blockDim.x;
  for (long i = tid; i < (long)EE * DD; i += stride)
    selW[i] = (__bf16)wsel[i];
  for (long i = tid; i < (long)EE * HH * DD; i += stride) {
    long e = i / ((long)HH * DD);
    long r = i % ((long)HH * DD);
    long h = r / DD, d = r % DD;                 // out index [e][h][d]
    keysT[i] = (__bf16)keys[(e * DD + d) * HH + h];
  }
  for (long i = tid; i < (long)EE * DD * HH; i += stride) {
    long e = i / ((long)DD * HH);
    long r = i % ((long)DD * HH);
    long nc = r / HH, h = r % HH;                // out index [e][n][h]
    valsT[i] = (__bf16)vals[(e * HH + h) * DD + nc];
  }
}

// ---------------------------------------------------------------------------
// Main kernel: one block = 32 tokens, 8 waves (wave32).
// Phase A: wave w owns scores h-tile w for BOTH 16-row M-subtiles (B reuse x2).
// Phase B: wave w owns output columns [w*128, w*128+128) for both subtiles.
// ---------------------------------------------------------------------------
__global__ __launch_bounds__(256) void moe_kernel(
    const float* __restrict__ x,
    const __bf16* __restrict__ selW,
    const __bf16* __restrict__ keysT,
    const __bf16* __restrict__ valsT,
    float* __restrict__ out) {
  __shared__ __attribute__((aligned(16))) __bf16 Xs[MTILE * XS_STRIDE];
  __shared__ __attribute__((aligned(16))) __bf16 Sc[MTILE * SC_STRIDE];
  __shared__ float sels[MTILE * 16];

  const int tid  = threadIdx.x;
  const int lane = tid & 31;
  const int wave = tid >> 5;
  const int n    = lane & 15;   // fragment column / A-row
  const int half = lane >> 4;   // lane-half selects K sub-range
  const long tok0 = (long)blockIdx.x * MTILE;

  // ---- Stage X tile (32 x 1024 fp32) into LDS as bf16 (coalesced float4)
  {
    const float4* xg = (const float4*)(x + tok0 * DD);
    for (int i = tid; i < MTILE * DD / 4; i += 256) {
      float4 v = xg[i];
      int m = (i * 4) / DD, c = (i * 4) % DD;
      __bf16* p = &Xs[m * XS_STRIDE + c];
      p[0] = (__bf16)v.x; p[1] = (__bf16)v.y;
      p[2] = (__bf16)v.z; p[3] = (__bf16)v.w;
    }
  }
  __syncthreads();

  // A fragment (16-bit A 16x32 layout). Lane L: row m = L%16. The 16 bf16
  // elements are K = kc+half*16 .. +15 contiguous (matches B fragment)
  // -> two ds_load_b128, no repack.
  auto loadA = [&](const __bf16* base, int stride, int kc) -> v16bf {
    union { v16bf v; uint4 q[2]; } a;
    const __bf16* row = base + n * stride + kc + half * 16;
    a.q[0] = *(const uint4*)(row);
    a.q[1] = *(const uint4*)(row + 8);
    return a.v;
  };

  // B fragment (16-bit B 32x16): lane L column n holds K = half*16..half*16+15,
  // contiguous in the transposed weight row -> one 32B global load.
  auto loadB = [&](const __bf16* rowptr) -> v16bf {
    union { v16bf v; uint4 q[2]; } b;
    const uint4* p = (const uint4*)(rowptr + half * 16);
    b.q[0] = p[0]; b.q[1] = p[1];
    return b.v;
  };

  // ---- Router: waves 0/1 compute sel = sigmoid(X @ Wsel^T), one 16x16 tile each
  if (wave < 2) {
    const __bf16* xrow = Xs + wave * 16 * XS_STRIDE;
    v8f sl = {};
    for (int kc = 0; kc < DD; kc += 32) {
      v16bf a = loadA(xrow, XS_STRIDE, kc);
      v16bf b = loadB(selW + (long)n * DD + kc);   // column n = expert n
      sl = __builtin_amdgcn_wmma_f32_16x16x32_bf16(
          false, a, false, b, (short)0, sl, false, false);
    }
#pragma unroll
    for (int v2 = 0; v2 < 8; ++v2) {
      int m = v2 + half * 8 + wave * 16;           // C layout: M = v + half*8
      sels[m * 16 + n] = __builtin_amdgcn_rcpf(1.0f + __expf(-sl[v2]));
    }
  }
  __syncthreads();

  // ---- Persistent output accumulators: 2 M-subtiles x 8 N-tiles per wave
  v8f acc0[8], acc1[8];
  v8f zero = {};
#pragma unroll
  for (int i = 0; i < 8; ++i) { acc0[i] = zero; acc1[i] = zero; }

  for (int e = 0; e < EE; ++e) {
    // Phase A: scores h-tile (2 x 16x16) = X @ K_e(:, wave*16+n)
    v8f sc0 = {}, sc1 = {};
    const __bf16* kb = keysT + ((long)e * HH + wave * 16 + n) * DD;
    for (int kc = 0; kc < DD; kc += 32) {
      v16bf b  = loadB(kb + kc);                       // shared by both subtiles
      v16bf a0 = loadA(Xs, XS_STRIDE, kc);
      v16bf a1 = loadA(Xs + 16 * XS_STRIDE, XS_STRIDE, kc);
      sc0 = __builtin_amdgcn_wmma_f32_16x16x32_bf16(
          false, a0, false, b, (short)0, sc0, false, false);
      sc1 = __builtin_amdgcn_wmma_f32_16x16x32_bf16(
          false, a1, false, b, (short)0, sc1, false, false);
    }
    // relu, fuse per-row sigmoid gate, write bf16 scores tiles to LDS
#pragma unroll
    for (int v2 = 0; v2 < 8; ++v2) {
      int m = v2 + half * 8;
      float s0 = sc0[v2];
      s0 = s0 > 0.0f ? s0 : 0.0f;
      s0 *= sels[m * 16 + e];
      Sc[m * SC_STRIDE + wave * 16 + n] = (__bf16)s0;
      float s1 = sc1[v2];
      s1 = s1 > 0.0f ? s1 : 0.0f;
      s1 *= sels[(m + 16) * 16 + e];
      Sc[(m + 16) * SC_STRIDE + wave * 16 + n] = (__bf16)s1;
    }
    __syncthreads();

    // Phase B: acc += scores(32x128) @ V_e(128, wave's 128 columns)
    v16bf af0[4], af1[4];
#pragma unroll
    for (int c = 0; c < 4; ++c) {
      af0[c] = loadA(Sc, SC_STRIDE, c * 32);
      af1[c] = loadA(Sc + 16 * SC_STRIDE, SC_STRIDE, c * 32);
    }
#pragma unroll
    for (int nt = 0; nt < 8; ++nt) {
      const __bf16* vb = valsT + ((long)e * DD + wave * 128 + nt * 16 + n) * HH;
#pragma unroll
      for (int c = 0; c < 4; ++c) {
        v16bf b = loadB(vb + c * 32);                  // shared by both subtiles
        acc0[nt] = __builtin_amdgcn_wmma_f32_16x16x32_bf16(
            false, af0[c], false, b, (short)0, acc0[nt], false, false);
        acc1[nt] = __builtin_amdgcn_wmma_f32_16x16x32_bf16(
            false, af1[c], false, b, (short)0, acc1[nt], false, false);
      }
    }
    __syncthreads();   // protect Sc before next expert overwrites it
  }

  // ---- Store output (fp32). Lanes 0-15 of a row are consecutive columns.
  float* og = out + tok0 * DD + wave * 128;
#pragma unroll
  for (int nt = 0; nt < 8; ++nt) {
#pragma unroll
    for (int v2 = 0; v2 < 8; ++v2) {
      int m = v2 + half * 8;
      og[(long)m * DD + nt * 16 + n] = acc0[nt][v2];
      og[(long)(m + 16) * DD + nt * 16 + n] = acc1[nt][v2];
    }
  }
}

// ---------------------------------------------------------------------------
extern "C" void kernel_launch(void* const* d_in, const int* in_sizes, int n_in,
                              void* d_out, int out_size, void* d_ws, size_t ws_size,
                              hipStream_t stream) {
  (void)in_sizes; (void)n_in; (void)out_size; (void)ws_size;
  const float* x    = (const float*)d_in[0];   // [B,S,D]
  const float* wsel = (const float*)d_in[1];   // [E,D]
  const float* keys = (const float*)d_in[2];   // [E,D,H]
  const float* vals = (const float*)d_in[3];   // [E,H,D]

  __bf16* selW  = (__bf16*)d_ws;                       // 16*1024
  __bf16* keysT = selW + (long)EE * DD;                // 16*128*1024
  __bf16* valsT = keysT + (long)EE * HH * DD;          // 16*1024*128
  // total ws use: (16K + 2M + 2M) bf16 ~ 8.4 MB

  prep_kernel<<<512, 256, 0, stream>>>(wsel, keys, vals, selW, keysT, valsT);
  moe_kernel<<<NTOK / MTILE, 256, 0, stream>>>(x, selW, keysT, valsT,
                                               (float*)d_out);
}
